// NonLocalBlock_58274116272402
// MI455X (gfx1250) — matrix operand order repaired and checked by reading, not compiled, not run on hardware
//
#include <hip/hip_runtime.h>
#include <math.h>

// ---------------------------------------------------------------------------
// NonLocalBlock for MI455X (gfx1250, wave32, WMMA f16 -> f32 accum)
//
// Shapes: x[B=32,C=1024,H=32,W=32], hp=256, N=H*W=1024, M=(H/2)*(W/2)=256.
//   q  = theta * x          : [256x1024]x[1024x1024] per batch
//   k,v= phi/g * maxpool(x) : [256x1024]x[1024x256]  per batch (pool fused)
//   attn: softmax(q^T k * hp^-1/2) @ v^T   (fused, scores stay in LDS)
//   out= z * vp + bias + x  : [1024x256]x[256x1024]  per batch
//
// All GEMMs: f16 operands staged in LDS in fragment-friendly [row][k32]
// layout, v_wmma_f32_16x16x32_f16, f32 accumulation.
// Workspace (f16): qT 16MB + kT 4MB + v 4MB + vpT 16MB = 40MB.
// ---------------------------------------------------------------------------

typedef __attribute__((ext_vector_type(16))) _Float16 v16h;
typedef __attribute__((ext_vector_type(8)))  _Float16 v8h;
typedef __attribute__((ext_vector_type(8)))  float    v8f;

#define BDIM 256

// A-matrix 16x32 f16 fragment (ISA 7.12.2): lanes 0-15 -> row=lane,
// K = 0..7 (v0..v3) and 16..23 (v4..v7); lanes 16-31 same rows, K+8.
__device__ __forceinline__ v16h frag_a(const _Float16* base, int row0, int ldk) {
  const int lane = threadIdx.x & 31;
  const int hs = lane >> 4, r = lane & 15;
  const _Float16* p = base + (size_t)(row0 + r) * ldk + 8 * hs;
  v8h lo = *(const v8h*)(p);        // K = 8*hs .. 8*hs+7
  v8h hi = *(const v8h*)(p + 16);   // K = 16+8*hs .. 16+8*hs+7
  v16h f;
#pragma unroll
  for (int j = 0; j < 8; ++j) { f[j] = lo[j]; f[8 + j] = hi[j]; }
  return f;
}

// B-matrix 32x16 f16 fragment: lanes 0-15 hold K=0..7,16..23, lanes 16-31
// hold K=8..15,24..31 (col = lane&15). Tile stored [n][k] (k contiguous).
__device__ __forceinline__ v16h frag_b(const _Float16* base, int row0, int ldk) {
  const int lane = threadIdx.x & 31;
  const int hs = lane >> 4, r = lane & 15;
  const _Float16* p = base + (size_t)(row0 + r) * ldk + 8 * hs;
  v8h lo = *(const v8h*)(p);
  v8h hi = *(const v8h*)(p + 16);
  v16h f;
#pragma unroll
  for (int j = 0; j < 8; ++j) { f[j] = lo[j]; f[8 + j] = hi[j]; }
  return f;
}

__device__ __forceinline__ v8f wmma_f16(v16h a, v16h b, v8f c) {
  return __builtin_amdgcn_wmma_f32_16x16x32_f16(
      /*neg_a=*/false, a, /*neg_b=*/false, b,
      /*c_mod=*/(short)0, c, /*reuse_a=*/false, /*reuse_b=*/false);
}

// ---------------------------------------------------------------------------
// Kernel 1: qT[b][n][d] = (theta_w @ x[b])^T + theta_b   (f16 out)
// WG = 256 d x 64 n, K=1024 in chunks of 32. 8 waves: wave w owns
// d-tiles {2w,2w+1} x 4 n-tiles = 8 accum tiles.
// ---------------------------------------------------------------------------
__global__ __launch_bounds__(BDIM)
void k_proj_q(const float* __restrict__ x, const float* __restrict__ tw,
              const float* __restrict__ tb, _Float16* __restrict__ qT)
{
  __shared__ _Float16 sA[256 * 32];   // theta chunk [d][c]
  __shared__ _Float16 sB[64 * 32];    // x chunk transposed -> [n][c]
  const int b   = blockIdx.y;
  const int n0  = blockIdx.x * 64;
  const int tid = threadIdx.x, wave = tid >> 5;
  const float* xb = x + (size_t)b * 1024 * 1024;

  v8f acc[2][4] = {};
  for (int kc = 0; kc < 1024; kc += 32) {
    if (kc + 32 < 1024)   // stream hint for next x chunk (global_prefetch_b8)
      __builtin_prefetch(xb + (size_t)(kc + 32) * 1024 + n0 + (tid & 63), 0, 1);
    for (int i = tid; i < 256 * 32; i += BDIM) {
      int r = i >> 5, k = i & 31;
      sA[i] = (_Float16)tw[(size_t)r * 1024 + kc + k];
    }
    for (int i = tid; i < 64 * 32; i += BDIM) {       // coalesced over n
      int n = i & 63, c = i >> 6;
      sB[n * 32 + c] = (_Float16)xb[(size_t)(kc + c) * 1024 + n0 + n];
    }
    __syncthreads();
    v16h bf[4];
#pragma unroll
    for (int nt = 0; nt < 4; ++nt) bf[nt] = frag_b(sB, nt * 16, 32);
#pragma unroll
    for (int dt = 0; dt < 2; ++dt) {
      v16h af = frag_a(sA, (wave * 2 + dt) * 16, 32);
#pragma unroll
      for (int nt = 0; nt < 4; ++nt)
        acc[dt][nt] = wmma_f16(af, bf[nt], acc[dt][nt]);
    }
    __syncthreads();
  }
  const int lane = tid & 31, col = lane & 15, rh = (lane >> 4) * 8;
#pragma unroll
  for (int dt = 0; dt < 2; ++dt)
#pragma unroll
    for (int nt = 0; nt < 4; ++nt)
#pragma unroll
      for (int r = 0; r < 8; ++r) {
        int d = (wave * 2 + dt) * 16 + rh + r;
        int n = n0 + nt * 16 + col;
        qT[((size_t)b * 1024 + n) * 256 + d] = (_Float16)(acc[dt][nt][r] + tb[d]);
      }
}

// ---------------------------------------------------------------------------
// Kernel 2: k = phi @ maxpool(x), v = g @ maxpool(x). Pool fused into the
// B-operand staging. z=blockIdx.z selects phi(0)/g(1).
// kT[b][m][d] (B operand of QK), v[b][d][m] (B operand of VP).
// ---------------------------------------------------------------------------
__global__ __launch_bounds__(BDIM)
void k_proj_kv(const float* __restrict__ x,
               const float* __restrict__ pw, const float* __restrict__ pb,
               const float* __restrict__ gw, const float* __restrict__ gb,
               _Float16* __restrict__ kT, _Float16* __restrict__ vH)
{
  __shared__ _Float16 sA[256 * 32];   // weight chunk [d][c]
  __shared__ _Float16 sB[64 * 32];    // pooled x chunk [m][c]
  const int b   = blockIdx.y;
  const int m0  = blockIdx.x * 64;
  const int z   = blockIdx.z;
  const float* w  = z ? gw : pw;
  const float* bi = z ? gb : pb;
  const float* xb = x + (size_t)b * 1024 * 1024;
  const int tid = threadIdx.x, wave = tid >> 5;

  v8f acc[2][4] = {};
  for (int kc = 0; kc < 1024; kc += 32) {
    for (int i = tid; i < 256 * 32; i += BDIM) {
      int r = i >> 5, k = i & 31;
      sA[i] = (_Float16)w[(size_t)r * 1024 + kc + k];
    }
    for (int i = tid; i < 64 * 32; i += BDIM) {       // fused 2x2 maxpool
      int m = i & 63, c = i >> 6;
      int gm = m0 + m, my = gm >> 4, mx = gm & 15;
      const float* px = xb + (size_t)(kc + c) * 1024 + my * 64 + mx * 2;
      float v01 = fmaxf(px[0], px[1]);
      float v23 = fmaxf(px[32], px[33]);
      sB[m * 32 + c] = (_Float16)fmaxf(v01, v23);
    }
    __syncthreads();
    v16h bf[4];
#pragma unroll
    for (int mt = 0; mt < 4; ++mt) bf[mt] = frag_b(sB, mt * 16, 32);
#pragma unroll
    for (int dt = 0; dt < 2; ++dt) {
      v16h af = frag_a(sA, (wave * 2 + dt) * 16, 32);
#pragma unroll
      for (int mt = 0; mt < 4; ++mt)
        acc[dt][mt] = wmma_f16(af, bf[mt], acc[dt][mt]);
    }
    __syncthreads();
  }
  const int lane = tid & 31, col = lane & 15, rh = (lane >> 4) * 8;
#pragma unroll
  for (int dt = 0; dt < 2; ++dt)
#pragma unroll
    for (int mt = 0; mt < 4; ++mt)
#pragma unroll
      for (int r = 0; r < 8; ++r) {
        int d = (wave * 2 + dt) * 16 + rh + r;
        int m = m0 + mt * 16 + col;
        float v = acc[dt][mt][r] + bi[d];
        if (z == 0) kT[((size_t)b * 256 + m) * 256 + d] = (_Float16)v;
        else        vH[((size_t)b * 256 + d) * 256 + m] = (_Float16)v;
      }
}

// ---------------------------------------------------------------------------
// Kernel 3: fused attention per (batch, 64-query tile).
//   S = (q^T k) * hp^-1/2  -> LDS f32 [64][256]
//   P = softmax_rows(S)    -> LDS f16 [64][256]
//   vpT[n][d] = P @ v^T    (K = m = 256)
// Dynamic LDS: 4KB sQ + 16KB sKV + 64KB sS + 32KB sP = 116KB (of 320KB/WGP).
// ---------------------------------------------------------------------------
__global__ __launch_bounds__(BDIM)
void k_attn(const _Float16* __restrict__ qT, const _Float16* __restrict__ kT,
            const _Float16* __restrict__ vH, _Float16* __restrict__ vpT)
{
  extern __shared__ char smem[];
  _Float16* sQ  = (_Float16*)smem;                            // [64][32]
  _Float16* sKV = (_Float16*)(smem + 4096);                   // [256][32]
  float*    sS  = (float*)   (smem + 4096 + 16384);           // [64][256]
  _Float16* sP  = (_Float16*)(smem + 4096 + 16384 + 65536);   // [64][256]

  const int b   = blockIdx.y;
  const int n0  = blockIdx.x * 64;
  const int tid = threadIdx.x, wave = tid >> 5, lane = tid & 31;
  const int col = lane & 15, rh = (lane >> 4) * 8;
  const int nt    = wave >> 1;          // n-tile 0..3 (16 rows each)
  const int half8 = (wave & 1) * 8;     // this wave's 8 m/d tiles

  const _Float16* qTb = qT + ((size_t)b * 1024 + n0) * 256;
  const _Float16* kTb = kT + (size_t)b * 256 * 256;
  const _Float16* vHb = vH + (size_t)b * 256 * 256;

  // ---- GEMM1: scores ----
  v8f acc[8] = {};
  for (int kc = 0; kc < 256; kc += 32) {
    for (int i = tid; i < 64 * 32; i += BDIM) {
      int r = i >> 5, k = i & 31;
      sQ[i] = qTb[(size_t)r * 256 + kc + k];
    }
    for (int i = tid; i < 256 * 32; i += BDIM) {
      int r = i >> 5, k = i & 31;
      sKV[i] = kTb[(size_t)r * 256 + kc + k];
    }
    __syncthreads();
    v16h af = frag_a(sQ, nt * 16, 32);
#pragma unroll
    for (int mt = 0; mt < 8; ++mt) {
      v16h bf = frag_b(sKV, (half8 + mt) * 16, 32);
      acc[mt] = wmma_f16(af, bf, acc[mt]);
    }
    __syncthreads();
  }
  const float scale = 0.0625f;          // hp^-1/2 = 1/16
#pragma unroll
  for (int mt = 0; mt < 8; ++mt)
#pragma unroll
    for (int r = 0; r < 8; ++r)
      sS[(nt * 16 + rh + r) * 256 + (half8 + mt) * 16 + col] = acc[mt][r] * scale;
  __syncthreads();

  // ---- row softmax (one thread per query row) ----
  if (tid < 64) {
    float* row = sS + tid * 256;
    float mx = row[0];
    for (int m = 1; m < 256; ++m) mx = fmaxf(mx, row[m]);
    float s = 0.f;
    for (int m = 0; m < 256; ++m) { float e = __expf(row[m] - mx); row[m] = e; s += e; }
    float inv = 1.f / s;
    _Float16* pr = sP + tid * 256;
    for (int m = 0; m < 256; ++m) pr[m] = (_Float16)(row[m] * inv);
  }
  __syncthreads();

  // ---- GEMM2: vpT = P @ v^T ----
  v8f acc2[8] = {};
  for (int mc = 0; mc < 256; mc += 32) {
    for (int i = tid; i < 256 * 32; i += BDIM) {
      int r = i >> 5, k = i & 31;
      sKV[i] = vHb[(size_t)r * 256 + mc + k];   // v stored [d][m]
    }
    __syncthreads();
    v16h af = frag_a(sP + mc, nt * 16, 256);
#pragma unroll
    for (int dt = 0; dt < 8; ++dt) {
      v16h bf = frag_b(sKV, (half8 + dt) * 16, 32);
      acc2[dt] = wmma_f16(af, bf, acc2[dt]);
    }
    __syncthreads();
  }
#pragma unroll
  for (int dt = 0; dt < 8; ++dt)
#pragma unroll
    for (int r = 0; r < 8; ++r) {
      int nr = nt * 16 + rh + r;
      int dc = (half8 + dt) * 16 + col;
      vpT[((size_t)b * 1024 + n0 + nr) * 256 + dc] = (_Float16)acc2[dt][r];
    }
}

// ---------------------------------------------------------------------------
// Kernel 4: out[b][c][n] = z_w @ vp + z_b + x. WG = 128 c x 64 n, K=256.
// ---------------------------------------------------------------------------
__global__ __launch_bounds__(BDIM)
void k_out(const float* __restrict__ x, const float* __restrict__ zw,
           const float* __restrict__ zb, const _Float16* __restrict__ vpT,
           float* __restrict__ out)
{
  __shared__ _Float16 sA[128 * 32];   // z_w chunk [c][d]
  __shared__ _Float16 sB[64 * 32];    // vpT chunk [n][d]
  const int b   = blockIdx.z;
  const int c0  = blockIdx.y * 128;
  const int n0  = blockIdx.x * 64;
  const int tid = threadIdx.x, wave = tid >> 5;

  v8f acc[4] = {};
  for (int kc = 0; kc < 256; kc += 32) {
    for (int i = tid; i < 128 * 32; i += BDIM) {
      int r = i >> 5, k = i & 31;
      sA[i] = (_Float16)zw[(size_t)(c0 + r) * 256 + kc + k];
    }
    for (int i = tid; i < 64 * 32; i += BDIM) {
      int r = i >> 5, k = i & 31;
      sB[i] = vpT[((size_t)b * 1024 + n0 + r) * 256 + kc + k];
    }
    __syncthreads();
    v16h af = frag_a(sA, wave * 16, 32);
#pragma unroll
    for (int ntt = 0; ntt < 4; ++ntt) {
      v16h bf = frag_b(sB, ntt * 16, 32);
      acc[ntt] = wmma_f16(af, bf, acc[ntt]);
    }
    __syncthreads();
  }
  const int lane = tid & 31, col = lane & 15, rh = (lane >> 4) * 8;
#pragma unroll
  for (int ntt = 0; ntt < 4; ++ntt)
#pragma unroll
    for (int r = 0; r < 8; ++r) {
      int c = c0 + wave * 16 + rh + r;
      int n = n0 + ntt * 16 + col;
      size_t idx = ((size_t)b * 1024 + c) * 1024 + n;
      out[idx] = acc[ntt][r] + zb[c] + x[idx];      // bias + residual
    }
}

// ---------------------------------------------------------------------------
extern "C" void kernel_launch(void* const* d_in, const int* in_sizes, int n_in,
                              void* d_out, int out_size, void* d_ws, size_t ws_size,
                              hipStream_t stream) {
  (void)in_sizes; (void)n_in; (void)out_size; (void)ws_size;
  const float* x  = (const float*)d_in[0];
  const float* tw = (const float*)d_in[1];
  const float* tb = (const float*)d_in[2];
  const float* pw = (const float*)d_in[3];
  const float* pb = (const float*)d_in[4];
  const float* gw = (const float*)d_in[5];
  const float* gb = (const float*)d_in[6];
  const float* zw = (const float*)d_in[7];
  const float* zb = (const float*)d_in[8];
  float* out = (float*)d_out;

  // workspace carve (f16): 40MB total
  _Float16* qT  = (_Float16*)d_ws;                    // [32][1024][256]
  _Float16* kT  = qT + (size_t)32 * 1024 * 256;       // [32][256][256]
  _Float16* vH  = kT + (size_t)32 * 256 * 256;        // [32][256][256]
  _Float16* vpT = vH + (size_t)32 * 256 * 256;        // [32][1024][256]

  k_proj_q <<<dim3(16, 32),    BDIM, 0, stream>>>(x, tw, tb, qT);
  k_proj_kv<<<dim3(4, 32, 2),  BDIM, 0, stream>>>(x, pw, pb, gw, gb, kT, vH);
  const size_t attn_smem = 4096 + 16384 + 65536 + 32768;   // 116KB
  k_attn   <<<dim3(16, 32),    BDIM, attn_smem, stream>>>(qT, kT, vH, vpT);
  k_out    <<<dim3(16, 8, 32), BDIM, 0, stream>>>(x, zw, zb, vpT, out);
}
